// LSTMModel_40492951666860
// MI455X (gfx1250) — matrix operand order repaired and checked by reading, not compiled, not run on hardware
//
#include <hip/hip_runtime.h>

// ---------------- types ----------------
typedef __attribute__((ext_vector_type(16))) __bf16 v16bf;
typedef __attribute__((ext_vector_type(8)))  __bf16 v8bf;
typedef __attribute__((ext_vector_type(8)))  float  v8f;
typedef unsigned short u16;

#define MAX_LEN 40
#define EMBED   50
#define HIDDEN  200
#define VOCAB   50000
#define BATCH   2048
#define NCOL    800          // 4*HIDDEN
#define KLSTM   256          // 250 padded -> 8 k-steps of 32
#define KPROJ   224          // 200 padded -> 7 k-steps of 32
#define CT_LSTM 50           // 800/16 col tiles
#define CT_PROJ 3125         // 50000/16 col tiles
#define PROJ_ROWS 64         // rows per projection workgroup (4 subtiles)

// ---------------- helpers ----------------
__device__ inline u16 f32_to_bf16(float f) {
    union { float f; unsigned u; } v; v.f = f;
    unsigned u = v.u;
    u += 0x7fffu + ((u >> 16) & 1u);   // round-to-nearest-even
    return (u16)(u >> 16);
}

// A fragment (16x32 bf16, CDNA5 layout): lane holds row M=lane&15,
// halves h -> K = ks*32 + h + (h>=8 ? 8:0) + (lane>=16 ? 8:0)
// i.e. two contiguous 8-half runs: [base .. base+7] and [base+16 .. base+23]
__device__ inline v16bf load_a_frag(const u16* base, int rowBase, int rowStrideH,
                                    int ks, int lane) {
    int row = rowBase + (lane & 15);
    int off = row * rowStrideH + ks * 32 + ((lane & 16) ? 8 : 0);
    v8bf lo = *(const v8bf*)(base + off);
    v8bf hi = *(const v8bf*)(base + off + 16);
    return __builtin_shufflevector(lo, hi, 0,1,2,3,4,5,6,7,8,9,10,11,12,13,14,15);
}

__device__ inline float fsigm(float x) { return 1.0f / (1.0f + __expf(-x)); }
__device__ inline float ftanh(float x) {
    x = fminf(fmaxf(x, -15.0f), 15.0f);
    float e = __expf(2.0f * x);
    return (e - 1.0f) / (e + 1.0f);
}

// ---------------- weight swizzle kernels ----------------
// B fragment layout (32x16 bf16): lane holds col N=lane&15,
// half h -> K = ks*32 + h + (h>=8 ? 8:0) + (lane>=16 ? 8:0).  Store the 16 halves a lane
// needs contiguously: Wb[ct][ks][lane][h] -> two global_load_b128 per frag.
__global__ void k_prep_wb(const float* __restrict__ W, u16* __restrict__ Wb) {
    int idx = blockIdx.x * blockDim.x + threadIdx.x;
    if (idx >= CT_LSTM * 8 * 32 * 16) return;
    int h    = idx & 15;
    int lane = (idx >> 4) & 31;
    int ks   = (idx >> 9) & 7;
    int ct   = idx >> 12;
    int n = ct * 16 + (lane & 15);
    int k = ks * 32 + h + ((h >= 8) ? 8 : 0) + ((lane >= 16) ? 8 : 0);
    float v = (k < EMBED + HIDDEN) ? W[k * NCOL + n] : 0.0f;
    Wb[idx] = f32_to_bf16(v);
}

__global__ void k_prep_ub(const float* __restrict__ U, u16* __restrict__ Ub) {
    int idx = blockIdx.x * blockDim.x + threadIdx.x;
    if (idx >= CT_PROJ * 7 * 32 * 16) return;
    int h    = idx & 15;
    int lane = (idx >> 4) & 31;
    int rem  = idx >> 9;
    int ks   = rem % 7;
    int ct   = rem / 7;
    int n = ct * 16 + (lane & 15);
    int k = ks * 32 + h + ((h >= 8) ? 8 : 0) + ((lane >= 16) ? 8 : 0);
    float v = (k < HIDDEN) ? U[(size_t)k * VOCAB + n] : 0.0f;
    Ub[idx] = f32_to_bf16(v);
}

// ---------------- persistent LSTM kernel ----------------
// 128 blocks x 256 threads; each block owns 16 batch rows and iterates all
// 40 timesteps with c/h resident in LDS (recurrence is row-independent, so
// no cross-workgroup sync is ever needed).
__global__ __launch_bounds__(256) void k_lstm(
        const int* __restrict__ tokens, const int* __restrict__ lengths,
        const float* __restrict__ emb,  const u16* __restrict__ Wb,
        const float* __restrict__ b_lstm, float* __restrict__ c_final) {
    __shared__ __align__(16) u16 ldsA[16 * KLSTM];   // [x_t | h] bf16, h persistent
    __shared__ float ldsZ[16 * NCOL];                // gate pre-activations
    __shared__ float ldsC[16 * HIDDEN];              // fp32 cell state
    __shared__ int tokLds[16];
    __shared__ int lenLds[16];

    const int tid  = threadIdx.x;
    const int rb   = blockIdx.x * 16;
    const int wave = tid >> 5;
    const int lane = tid & 31;

    // init: zero A (h region + K pad) and c state
    for (int e = tid; e < 16 * KLSTM; e += 256) ldsA[e] = 0;
    for (int e = tid; e < 16 * HIDDEN; e += 256) ldsC[e] = 0.0f;
    if (tid < 16) lenLds[tid] = lengths[rb + tid];
    __syncthreads();

    for (int t = 0; t < MAX_LEN; ++t) {
        if (tid < 16) tokLds[tid] = tokens[(rb + tid) * MAX_LEN + t];
        __syncthreads();
        // stage embeddings (k < 50) as bf16
        for (int e = tid; e < 16 * EMBED; e += 256) {
            int r = e / EMBED, k = e % EMBED;
            ldsA[r * KLSTM + k] = f32_to_bf16(emb[tokLds[r] * EMBED + k]);
        }
        __syncthreads();

        // z = [x_t, h] @ W  (16 x 800, K=256); A frags re-read from LDS per column tile
        for (int ct = wave; ct < CT_LSTM; ct += 8) {
            v8f acc = {};
            const u16* wbt = Wb + ct * (8 * 512);
            #pragma unroll
            for (int ks = 0; ks < 8; ++ks) {
                v16bf a = load_a_frag(ldsA, 0, KLSTM, ks, lane);
                v16bf b = *(const v16bf*)(wbt + ks * 512 + lane * 16);
                acc = __builtin_amdgcn_wmma_f32_16x16x32_bf16(
                          false, a, false, b, (short)0, acc, false, false);
            }
            int n = ct * 16 + (lane & 15);
            int mBase = (lane >= 16) ? 8 : 0;
            #pragma unroll
            for (int r = 0; r < 8; ++r) ldsZ[(mBase + r) * NCOL + n] = acc[r];
        }
        __syncthreads();

        // gates (i, j, f, o), TF LSTMCell semantics + sequence-length mask
        for (int e = tid; e < 16 * HIDDEN; e += 256) {
            int m = e / HIDDEN, hc = e % HIDDEN;
            if (t < lenLds[m]) {
                float iv = ldsZ[m * NCOL + hc]              + b_lstm[hc];
                float jv = ldsZ[m * NCOL + HIDDEN + hc]     + b_lstm[HIDDEN + hc];
                float fv = ldsZ[m * NCOL + 2 * HIDDEN + hc] + b_lstm[2 * HIDDEN + hc];
                float ov = ldsZ[m * NCOL + 3 * HIDDEN + hc] + b_lstm[3 * HIDDEN + hc];
                float cn = fsigm(fv + 1.0f) * ldsC[e] + fsigm(iv) * ftanh(jv);
                float hn = fsigm(ov) * ftanh(cn);
                ldsC[e] = cn;
                ldsA[m * KLSTM + EMBED + hc] = f32_to_bf16(hn);
            }
        }
        __syncthreads();
    }

    for (int e = tid; e < 16 * HIDDEN; e += 256)
        c_final[(rb + e / HIDDEN) * HIDDEN + (e % HIDDEN)] = ldsC[e];
}

// ---------------- projection: logits = c_final @ U + b2 ----------------
// 64 rows per workgroup: each B fragment fetched from L2 feeds 4 WMMAs,
// cutting Ub L2 traffic 4x (below the 410 MB HBM output-write cost).
__global__ __launch_bounds__(256) void k_proj(
        const float* __restrict__ c_final, const u16* __restrict__ Ub,
        const float* __restrict__ b2, float* __restrict__ out) {
    __shared__ __align__(16) u16 ldsA[PROJ_ROWS * KPROJ];
    const int tid = threadIdx.x;
    const int rb  = blockIdx.y * PROJ_ROWS;
    for (int e = tid; e < PROJ_ROWS * KPROJ; e += 256) {
        int r = e / KPROJ, k = e % KPROJ;
        ldsA[e] = (k < HIDDEN) ? f32_to_bf16(c_final[(rb + r) * HIDDEN + k]) : (u16)0;
    }
    __syncthreads();
    const int wave = tid >> 5, lane = tid & 31;
    int ct = blockIdx.x * 8 + wave;
    if (ct >= CT_PROJ) return;

    v8f acc[4] = {};
    const u16* ubt = Ub + (size_t)ct * (7 * 512);
    #pragma unroll
    for (int ks = 0; ks < 7; ++ks) {
        v16bf b = *(const v16bf*)(ubt + ks * 512 + lane * 16);
        #pragma unroll
        for (int rs = 0; rs < 4; ++rs) {
            v16bf a = load_a_frag(ldsA, rs * 16, KPROJ, ks, lane);
            acc[rs] = __builtin_amdgcn_wmma_f32_16x16x32_bf16(
                          false, a, false, b, (short)0, acc[rs], false, false);
        }
    }
    int n = ct * 16 + (lane & 15);
    float bias = b2[n];
    int mBase = (lane >= 16) ? 8 : 0;
    #pragma unroll
    for (int rs = 0; rs < 4; ++rs)
        #pragma unroll
        for (int r = 0; r < 8; ++r)
            out[(size_t)(rb + rs * 16 + mBase + r) * VOCAB + n] = acc[rs][r] + bias;
}

// ---------------- softmax ----------------
__global__ __launch_bounds__(256) void k_softmax_reduce(
        const float* __restrict__ logits, float* __restrict__ row_max,
        float* __restrict__ row_sum) {
    const int row = blockIdx.x, tid = threadIdx.x;
    const float* x = logits + (size_t)row * VOCAB;
    float m = -3.4e38f, s = 0.0f;                 // online max/sum, single pass
    for (int i = tid; i < VOCAB; i += 256) {
        float v = x[i];
        if (v > m) { s = s * __expf(m - v) + 1.0f; m = v; }
        else       { s += __expf(v - m); }
    }
    __shared__ float sm[256], ss[256];
    sm[tid] = m; ss[tid] = s;
    __syncthreads();
    for (int off = 128; off > 0; off >>= 1) {
        if (tid < off) {
            float m2 = sm[tid + off], s2 = ss[tid + off];
            float M = fmaxf(sm[tid], m2);
            ss[tid] = ss[tid] * __expf(sm[tid] - M) + s2 * __expf(m2 - M);
            sm[tid] = M;
        }
        __syncthreads();
    }
    if (tid == 0) { row_max[row] = sm[0]; row_sum[row] = ss[0]; }
}

__global__ void k_softmax_norm(float* __restrict__ out,
                               const float* __restrict__ row_max,
                               const float* __restrict__ row_sum) {
    const size_t total4 = (size_t)BATCH * VOCAB / 4;   // VOCAB % 4 == 0
    for (size_t i = (size_t)blockIdx.x * blockDim.x + threadIdx.x;
         i < total4; i += (size_t)gridDim.x * blockDim.x) {
        size_t e = i * 4;
        int row = (int)(e / VOCAB);
        float m = row_max[row];
        float inv = 1.0f / row_sum[row];
        float4 v = ((float4*)out)[i];
        v.x = __expf(v.x - m) * inv;
        v.y = __expf(v.y - m) * inv;
        v.z = __expf(v.z - m) * inv;
        v.w = __expf(v.w - m) * inv;
        ((float4*)out)[i] = v;
    }
}

// ---------------- launch ----------------
extern "C" void kernel_launch(void* const* d_in, const int* in_sizes, int n_in,
                              void* d_out, int out_size, void* d_ws, size_t ws_size,
                              hipStream_t stream) {
    const int*   tokens  = (const int*)  d_in[0];
    const int*   lengths = (const int*)  d_in[1];
    const float* emb     = (const float*)d_in[2];
    const float* W_lstm  = (const float*)d_in[3];
    const float* b_lstm  = (const float*)d_in[4];
    const float* U       = (const float*)d_in[5];
    const float* b2      = (const float*)d_in[6];
    float* out = (float*)d_out;

    char* ws = (char*)d_ws;
    float* c_final = (float*)(ws);                          // 2048*200*4      = 1,638,400
    u16*   Wb      = (u16*)  (ws + 1638400);                // 50*8*512*2      =   409,600
    u16*   Ub      = (u16*)  (ws + 2048000);                // 3125*7*512*2    = 22,400,000
    float* row_max = (float*)(ws + 24448000);               // 2048*4
    float* row_sum = (float*)(ws + 24456192);               // 2048*4

    k_prep_wb<<<(CT_LSTM * 8 * 512 + 255) / 256, 256, 0, stream>>>(W_lstm, Wb);
    k_prep_ub<<<(CT_PROJ * 7 * 512 + 255) / 256, 256, 0, stream>>>(U, Ub);

    k_lstm<<<BATCH / 16, 256, 0, stream>>>(tokens, lengths, emb, Wb, b_lstm, c_final);

    dim3 pg((CT_PROJ + 7) / 8, BATCH / PROJ_ROWS);
    k_proj<<<pg, 256, 0, stream>>>(c_final, Ub, b2, out);

    k_softmax_reduce<<<BATCH, 256, 0, stream>>>(out, row_max, row_sum);
    k_softmax_norm<<<8192, 256, 0, stream>>>(out, row_max, row_sum);
}
